// LSTM_15556371546189
// MI455X (gfx1250) — compile-verified, matrix operand order checked by reading, not run on hardware
//
#include <hip/hip_runtime.h>

// ---------------------------------------------------------------------------
// LSTM LM on MI455X (gfx1250)
//   - all GEMM operands pre-converted to bf16 (f32 accumulate via WMMA)
//   - GEMM B-panels staged into LDS by the Tensor Data Mover (double-buffered)
//   - persistent recurrent kernel with grid barrier between timesteps
// Sizes fixed by the reference: V=32000,E=512,H=1024,O=32000,B=16,T=256
// ---------------------------------------------------------------------------

#define E_DIM   512
#define H_DIM   1024
#define G4      4096      // 4*H
#define B_DIM   16
#define T_DIM   256
#define BT      4096      // B*T
#define O_DIM   32000
#define V_DIM   32000
#define NWG     16        // workgroups in persistent recurrent kernel

typedef __attribute__((ext_vector_type(16))) __bf16    bf16x16;
typedef __attribute__((ext_vector_type(8)))  __bf16    bf16x8;
typedef __attribute__((ext_vector_type(4)))  __bf16    bf16x4;
typedef __attribute__((ext_vector_type(8)))  float     f32x8;
typedef __attribute__((ext_vector_type(4)))  unsigned  u32x4;
typedef __attribute__((ext_vector_type(8)))  int       i32x8;
typedef __attribute__((ext_vector_type(4)))  int       i32x4;

__device__ __forceinline__ __bf16 f2bf(float f) { return (__bf16)f; }

__device__ __forceinline__ float fsig(float x) {
  return 1.0f / (1.0f + __expf(-x));
}

// A/B fragment from 32 contiguous bf16 K-values (LDS or global pointer).
// ISA 7.12.2 (16-bit 16x32): lanes<16 hold K{0..7,16..23}, lanes>=16 hold
// K{8..15,24..31}, packed in K order.
__device__ __forceinline__ bf16x16 frag_k32(const __bf16* row, int lane) {
  const int h8 = (lane >> 4) << 3;
  bf16x8 lo = *(const bf16x8*)(row + h8);
  bf16x8 hi = *(const bf16x8*)(row + 16 + h8);
  bf16x16 r;
#pragma unroll
  for (int i = 0; i < 8; ++i) { r[i] = lo[i]; r[8 + i] = hi[i]; }
  return r;
}

__device__ __forceinline__ f32x8 wmma_bf16(bf16x16 a, bf16x16 b, f32x8 c) {
  return __builtin_amdgcn_wmma_f32_16x16x32_bf16(false, a, false, b,
                                                 (short)0, c, false, false);
}

// ---------------------------------------------------------------------------
// Tensor Data Mover: async 2D tile load (bf16) global -> LDS.
// D# per CDNA5 ISA:
//   g0: [1:0]count=1 | lds_addr | global_addr[56:0] | type=2
//   g1: data_size=1(2B) | tensor_dim0/1 | tile_dim0=32, tile_dim1=rows |
//       tensor_dim0_stride=K
// ---------------------------------------------------------------------------
__device__ __forceinline__ void tdm_load_2d_bf16(const __bf16* gptr,
                                                 unsigned lds_off,
                                                 int K, int rows) {
  unsigned long long ga = (unsigned long long)(uintptr_t)gptr;
  u32x4 g0 = { 1u,                                    // count=1, user mode
               lds_off,                               // LDS byte offset
               (unsigned)(ga & 0xFFFFFFFFu),
               (unsigned)((ga >> 32) & 0x01FFFFFFu) | (2u << 30) }; // type=2
  unsigned k = (unsigned)K;
  unsigned td1 = 1u << 20;                            // generous OOB bound
  i32x8 g1 = { (int)(1u << 16),                       // data_size = 2 bytes
               (int)((k & 0xFFFFu) << 16),            // tensor_dim0 lo
               (int)((k >> 16) | ((td1 & 0xFFFFu) << 16)),
               (int)((td1 >> 16) | (32u << 16)),      // tile_dim0 = 32
               (int)(unsigned)rows,                   // tile_dim1, tile_dim2=0
               (int)k,                                // tensor_dim0_stride lo
               0, 0 };
  i32x4 z4 = { 0, 0, 0, 0 };
#if defined(__clang_major__) && (__clang_major__ >= 23)
  i32x8 z8 = { 0, 0, 0, 0, 0, 0, 0, 0 };
  __builtin_amdgcn_tensor_load_to_lds(g0, g1, z4, z4, z8, 0);
#else
  __builtin_amdgcn_tensor_load_to_lds(g0, g1, z4, z4, 0);
#endif
}

// ---------------------------------------------------------------------------
// GEMM: out[M,N] = A[M,K](bf16) * W[N,K](bf16)^T + bias[N]   (f32 out)
// A rows optionally gathered from bf16 `emb` via `tokens`.
// 128x128 block tile; A staged by threads, B staged by TDM; double-buffered.
// ---------------------------------------------------------------------------
__global__ __launch_bounds__(256, 2)
void gemm_bf16(const __bf16* __restrict__ A,
               const int*    __restrict__ tokens,
               const __bf16* __restrict__ emb,
               const __bf16* __restrict__ W,
               const float*  __restrict__ bias,
               float* __restrict__ out,
               int K, int ldo)
{
  __shared__ __bf16 As[2][128][32];   // 2 x 8 KB
  __shared__ __bf16 Bs[2][128][32];   // 2 x 8 KB (TDM destination)

  const int tid  = threadIdx.x;
  const int lane = tid & 31;
  const int wave = tid >> 5;
  const int bm   = blockIdx.y;
  const int bn   = blockIdx.x;

  const int lrow = tid >> 1;
  const int lcol = (tid & 1) << 4;
  const int arow = bm * 128 + lrow;
  const __bf16* arowp = tokens ? (emb + (size_t)tokens[arow] * K)
                               : (A   + (size_t)arow * K);
  const __bf16* wpanel = W + (size_t)(bn * 128) * K;  // 128 N-rows, K-major

  const unsigned ldsB0 = (unsigned)(uintptr_t)&Bs[0][0][0];
  const unsigned ldsB1 = (unsigned)(uintptr_t)&Bs[1][0][0];

  const int nlo = lane & 15;
  const int hh  = lane >> 4;
  const int mw  = wave * 16;

  f32x8 acc[8];
#pragma unroll
  for (int nt = 0; nt < 8; ++nt) {
    float bv = bias ? bias[bn * 128 + nt * 16 + nlo] : 0.0f;
#pragma unroll
    for (int r = 0; r < 8; ++r) acc[nt][r] = bv;
  }

  // prologue: stage k0 = 0 panels
  {
    bf16x8 v0 = *(const bf16x8*)(arowp + lcol);
    bf16x8 v1 = *(const bf16x8*)(arowp + lcol + 8);
    *(bf16x8*)&As[0][lrow][lcol]     = v0;
    *(bf16x8*)&As[0][lrow][lcol + 8] = v1;
  }
  if (wave == 0) tdm_load_2d_bf16(wpanel, ldsB0, K, 128);

  int buf = 0;
  for (int k0 = 0; k0 < K; k0 += 32) {
    if (wave == 0) __builtin_amdgcn_s_wait_tensorcnt(0);
    __syncthreads();   // panel `buf` (A via DS, B via TDM) is ready

    if (k0 + 32 < K) { // stage next panels into buf^1 while computing
      bf16x8 v0 = *(const bf16x8*)(arowp + k0 + 32 + lcol);
      bf16x8 v1 = *(const bf16x8*)(arowp + k0 + 32 + lcol + 8);
      *(bf16x8*)&As[buf ^ 1][lrow][lcol]     = v0;
      *(bf16x8*)&As[buf ^ 1][lrow][lcol + 8] = v1;
      if (wave == 0)
        tdm_load_2d_bf16(wpanel + k0 + 32, (buf ? ldsB0 : ldsB1), K, 128);
    }

    // preload all fragments, then issue WMMAs back-to-back
    bf16x16 af = frag_k32(&As[buf][mw + nlo][0], lane);
    bf16x16 bfr[8];
#pragma unroll
    for (int nt = 0; nt < 8; ++nt)
      bfr[nt] = frag_k32(&Bs[buf][nt * 16 + nlo][0], lane);
#pragma unroll
    for (int nt = 0; nt < 8; ++nt)
      acc[nt] = wmma_bf16(af, bfr[nt], acc[nt]);
    buf ^= 1;
  }

  // C/D layout: vgpr r <-> M = (lane>=16 ? 8 : 0)+r, N = lane&15
#pragma unroll
  for (int nt = 0; nt < 8; ++nt) {
    const int col = bn * 128 + nt * 16 + nlo;
#pragma unroll
    for (int r = 0; r < 8; ++r) {
      const int row = bm * 128 + mw + hh * 8 + r;
      out[(size_t)row * ldo + col] = acc[nt][r];
    }
  }
}

// ---------------------------------------------------------------------------
// Persistent LSTM recurrence for one layer (bf16 weights/h, f32 state).
// WG `wg` owns hidden columns [wg*64, wg*64+64): all 4 gate blocks local.
// ---------------------------------------------------------------------------
__global__ __launch_bounds__(256, 1)
void lstm_recurrent(const float*  __restrict__ xp,    // [BT,4H] f32 x-proj (+bias)
                    const __bf16* __restrict__ Whh,   // [4H,H] bf16
                    float*  __restrict__ cbuf,        // [B,H] f32
                    __bf16* __restrict__ hb0,         // [B,H] bf16 double buffer
                    __bf16* __restrict__ hb1,
                    __bf16* __restrict__ hseq,        // [BT,H] bf16 layer output
                    unsigned* __restrict__ counter)
{
  __shared__ __bf16 Hs[B_DIM][H_DIM];   // 32 KB staged h_{t-1}
  __shared__ float  Gs[4][B_DIM][64];   // 16 KB gate slices

  const int tid  = threadIdx.x;
  const int lane = tid & 31;
  const int wave = tid >> 5;
  const int wg   = blockIdx.x;
  const int nlo  = lane & 15;
  const int hh   = lane >> 4;

  // 16 output tiles (4 gates x 4 jtiles); 8 waves -> 2 tiles each
  const int tA = wave * 2, tB = tA + 1;
  const int gA = tA >> 2, jA = tA & 3;
  const int gB = tB >> 2, jB = tB & 3;
  const int n0 = gA * H_DIM + wg * 64 + jA * 16 + nlo;
  const int n1 = gB * H_DIM + wg * 64 + jB * 16 + nlo;
  const __bf16* w0 = Whh + (size_t)n0 * H_DIM;
  const __bf16* w1 = Whh + (size_t)n1 * H_DIM;

  for (int t = 0; t < T_DIM; ++t) {
    const __bf16* hprev = (t & 1) ? hb1 : hb0;
    __bf16*       hnext = (t & 1) ? hb0 : hb1;

    // stage h_{t-1} -> LDS (pure bf16 copy, 128B per thread)
    {
      const int base = tid * 64;
      __bf16* dst = &Hs[0][0];
#pragma unroll
      for (int i = 0; i < 8; ++i) {
        bf16x8 v = *(const bf16x8*)(hprev + base + i * 8);
        *(bf16x8*)(dst + base + i * 8) = v;
      }
    }
    __syncthreads();

    // init accumulators with precomputed input projection (includes bias)
    f32x8 a0, a1;
#pragma unroll
    for (int r = 0; r < 8; ++r) {
      const int m = hh * 8 + r;
      a0[r] = xp[(size_t)(m * T_DIM + t) * G4 + n0];
      a1[r] = xp[(size_t)(m * T_DIM + t) * G4 + n1];
    }

    // gates += h_{t-1} @ Whh^T  (K = 1024; W streamed bf16 from L2)
#pragma unroll 4
    for (int k0 = 0; k0 < H_DIM; k0 += 32) {
      __builtin_prefetch(w0 + k0 + 512, 0, 0);
      __builtin_prefetch(w1 + k0 + 512, 0, 0);
      bf16x16 af = frag_k32(&Hs[nlo][k0], lane);
      bf16x16 b0 = frag_k32(w0 + k0, lane);
      bf16x16 b1 = frag_k32(w1 + k0, lane);
      a0 = wmma_bf16(af, b0, a0);
      a1 = wmma_bf16(af, b1, a1);
    }

    // gate slices -> LDS for the cross-gate combine
#pragma unroll
    for (int r = 0; r < 8; ++r) {
      const int m = hh * 8 + r;
      Gs[gA][m][jA * 16 + nlo] = a0[r];
      Gs[gB][m][jB * 16 + nlo] = a1[r];
    }
    __syncthreads();

    // c = sig(f)*c + sig(i)*tanh(g);  h = sig(o)*tanh(c)   (vectorized x4)
    {
      const int m  = tid >> 4;
      const int j0 = (tid & 15) * 4;
      const int colb = wg * 64 + j0;
      float4 gi = *(const float4*)&Gs[0][m][j0];
      float4 gf = *(const float4*)&Gs[1][m][j0];
      float4 gg = *(const float4*)&Gs[2][m][j0];
      float4 go = *(const float4*)&Gs[3][m][j0];
      float4 cv = *(const float4*)&cbuf[m * H_DIM + colb];
      float hv[4];
      {
        float c;
        c = fsig(gf.x) * cv.x + fsig(gi.x) * tanhf(gg.x); cv.x = c;
        hv[0] = fsig(go.x) * tanhf(c);
        c = fsig(gf.y) * cv.y + fsig(gi.y) * tanhf(gg.y); cv.y = c;
        hv[1] = fsig(go.y) * tanhf(c);
        c = fsig(gf.z) * cv.z + fsig(gi.z) * tanhf(gg.z); cv.z = c;
        hv[2] = fsig(go.z) * tanhf(c);
        c = fsig(gf.w) * cv.w + fsig(gi.w) * tanhf(gg.w); cv.w = c;
        hv[3] = fsig(go.w) * tanhf(c);
      }
      *(float4*)&cbuf[m * H_DIM + colb] = cv;
      bf16x4 hb;
      hb[0] = f2bf(hv[0]); hb[1] = f2bf(hv[1]);
      hb[2] = f2bf(hv[2]); hb[3] = f2bf(hv[3]);
      *(bf16x4*)&hnext[m * H_DIM + colb] = hb;
      *(bf16x4*)&hseq[(size_t)(m * T_DIM + t) * H_DIM + colb] = hb;
    }

    // grid-wide step barrier (monotone counter; target = NWG*(t+1))
    __threadfence();
    __syncthreads();
    if (tid == 0) {
      __hip_atomic_fetch_add(counter, 1u, __ATOMIC_ACQ_REL,
                             __HIP_MEMORY_SCOPE_AGENT);
      const unsigned target = (unsigned)(NWG * (t + 1));
      while (__hip_atomic_load(counter, __ATOMIC_ACQUIRE,
                               __HIP_MEMORY_SCOPE_AGENT) < target)
        __builtin_amdgcn_s_sleep(2);
    }
    __syncthreads();
  }
}

// fp32 -> bf16 bulk converter (8 elements / thread, b128 in, b128 out)
__global__ __launch_bounds__(256)
void cvt_f32_bf16(const float* __restrict__ src, __bf16* __restrict__ dst,
                  size_t n) {
  const size_t i = ((size_t)blockIdx.x * 256 + threadIdx.x) * 8;
  if (i >= n) return;
  float4 a = *(const float4*)(src + i);
  float4 b = *(const float4*)(src + i + 4);
  bf16x8 o;
  o[0] = f2bf(a.x); o[1] = f2bf(a.y); o[2] = f2bf(a.z); o[3] = f2bf(a.w);
  o[4] = f2bf(b.x); o[5] = f2bf(b.y); o[6] = f2bf(b.z); o[7] = f2bf(b.w);
  *(bf16x8*)(dst + i) = o;
}

__global__ void zero_f32(float* __restrict__ p, int n) {
  const int i = blockIdx.x * blockDim.x + threadIdx.x;
  if (i < n) p[i] = 0.0f;
}

// ---------------------------------------------------------------------------
extern "C" void kernel_launch(void* const* d_in, const int* in_sizes, int n_in,
                              void* d_out, int out_size, void* d_ws, size_t ws_size,
                              hipStream_t stream) {
  (void)in_sizes; (void)n_in; (void)out_size; (void)ws_size;

  const int*   tokens = (const int*)  d_in[0];
  const float* emb    = (const float*)d_in[1];
  const float* W_ih0  = (const float*)d_in[2];
  const float* W_hh0  = (const float*)d_in[3];
  const float* b0     = (const float*)d_in[4];
  const float* W_ih1  = (const float*)d_in[5];
  const float* W_hh1  = (const float*)d_in[6];
  const float* b1     = (const float*)d_in[7];
  const float* W_fc   = (const float*)d_in[8];
  const float* b_fc   = (const float*)d_in[9];
  float* out = (float*)d_out;

  // --- workspace layout (256B-aligned carve-outs) ---
  char* p = (char*)d_ws;
  auto alloc = [&](size_t bytes) -> void* {
    void* r = (void*)p;
    p += (bytes + 255) & ~(size_t)255;
    return r;
  };
  __bf16* hbufs  = (__bf16*)alloc((size_t)4 * B_DIM * H_DIM * 2); // h dbl bufs x2 layers
  float*  cbufs  = (float*) alloc((size_t)2 * B_DIM * H_DIM * 4); // c per layer
  unsigned* counters = (unsigned*)alloc(256);
  char* zend = p;
  float*  xp0    = (float*) alloc((size_t)BT * G4 * 4);           // 64 MB
  float*  xp1    = (float*) alloc((size_t)BT * G4 * 4);           // 64 MB
  __bf16* hseq0  = (__bf16*)alloc((size_t)BT * H_DIM * 2);        //  8 MB
  __bf16* hseq1  = (__bf16*)alloc((size_t)BT * H_DIM * 2);        //  8 MB
  __bf16* embB   = (__bf16*)alloc((size_t)V_DIM * E_DIM * 2);     // 32 MB
  __bf16* Wih0b  = (__bf16*)alloc((size_t)G4 * E_DIM * 2);        //  4 MB
  __bf16* Whh0b  = (__bf16*)alloc((size_t)G4 * H_DIM * 2);        //  8 MB
  __bf16* Wih1b  = (__bf16*)alloc((size_t)G4 * H_DIM * 2);        //  8 MB
  __bf16* Whh1b  = (__bf16*)alloc((size_t)G4 * H_DIM * 2);        //  8 MB
  __bf16* Wfcb   = (__bf16*)alloc((size_t)O_DIM * H_DIM * 2);     // 64 MB

  __bf16* h0a = hbufs;
  __bf16* h0b = h0a + B_DIM * H_DIM;
  __bf16* h1a = h0b + B_DIM * H_DIM;
  __bf16* h1b = h1a + B_DIM * H_DIM;
  float*  c0  = cbufs;
  float*  c1  = c0 + B_DIM * H_DIM;

  // zero recurrent state + barrier counters (every call: graph-replay safe)
  const int zn = (int)(((char*)zend - (char*)hbufs) / 4);
  zero_f32<<<(zn + 255) / 256, 256, 0, stream>>>((float*)hbufs, zn);

  // bulk fp32 -> bf16 weight/embedding conversion
  auto cvt = [&](const float* s, __bf16* d, size_t n) {
    cvt_f32_bf16<<<(unsigned)((n / 8 + 255) / 256), 256, 0, stream>>>(s, d, n);
  };
  cvt(emb,   embB,  (size_t)V_DIM * E_DIM);
  cvt(W_ih0, Wih0b, (size_t)G4 * E_DIM);
  cvt(W_hh0, Whh0b, (size_t)G4 * H_DIM);
  cvt(W_ih1, Wih1b, (size_t)G4 * H_DIM);
  cvt(W_hh1, Whh1b, (size_t)G4 * H_DIM);
  cvt(W_fc,  Wfcb,  (size_t)O_DIM * H_DIM);

  // layer 0 input projection (embedding gather fused into A load)
  gemm_bf16<<<dim3(G4 / 128, BT / 128), 256, 0, stream>>>(
      nullptr, tokens, embB, Wih0b, b0, xp0, E_DIM, G4);

  lstm_recurrent<<<NWG, 256, 0, stream>>>(xp0, Whh0b, c0, h0a, h0b, hseq0,
                                          counters + 0);

  // layer 1 input projection
  gemm_bf16<<<dim3(G4 / 128, BT / 128), 256, 0, stream>>>(
      hseq0, nullptr, nullptr, Wih1b, b1, xp1, H_DIM, G4);

  lstm_recurrent<<<NWG, 256, 0, stream>>>(xp1, Whh1b, c1, h1a, h1b, hseq1,
                                          counters + 4);

  // final FC -> logits (f32)
  gemm_bf16<<<dim3(O_DIM / 128, BT / 128), 256, 0, stream>>>(
      hseq1, nullptr, nullptr, Wfcb, b_fc, out, H_DIM, O_DIM);
}
